// EdgeGATModel_73272142070365
// MI455X (gfx1250) — matrix operand order repaired and checked by reading, not compiled, not run on hardware
//
#include <hip/hip_runtime.h>
#include <hip/hip_bf16.h>
#include <math.h>

typedef __attribute__((ext_vector_type(16))) _Float16 v16h;
typedef __attribute__((ext_vector_type(8)))  _Float16 v8h;
typedef __attribute__((ext_vector_type(8)))  float    v8f;

#define BN_EPS    1e-5f
#define NEG_SLOPE 0.2f

#define WMMA_F16(a, b, c) \
  __builtin_amdgcn_wmma_f32_16x16x32_f16(false, (a), false, (b), (short)0, (c), false, false)

__device__ __forceinline__ float elu_f(float v)   { return v > 0.f ? v : __expf(v) - 1.f; }
__device__ __forceinline__ float lrelu_f(float v) { return v > 0.f ? v : NEG_SLOPE * v; }

// float atomic max via ordered-int trick (init to -inf)
__device__ __forceinline__ void atomic_max_f(float* addr, float val) {
  if (val >= 0.f) atomicMax((int*)addr, __float_as_int(val));
  else            atomicMin((unsigned int*)addr, __float_as_uint(val));
}

// Load a per-lane WMMA fragment: two 16-byte chunks at p and p+gap halves.
// Works for both global and LDS pointers (addrspace inferred after inlining).
template <typename P>
__device__ __forceinline__ v16h ld_frag(P p, int gap) {
  union { v16h v; v8h h[2]; } u;
  u.h[0] = *(const v8h*)(p);
  u.h[1] = *(const v8h*)(p + gap);
  return u.v;
}

// Cooperatively stage a weight matrix (global f16) into LDS, 16B chunks.
__device__ __forceinline__ void stage_w(const _Float16* __restrict__ g,
                                        _Float16* s, int n) {
  for (int i = threadIdx.x * 8; i < n; i += blockDim.x * 8)
    *(v8h*)(s + i) = *(const v8h*)(g + i);
  __syncthreads();
}

__device__ __forceinline__ void get_st(const int* src, const int* dst, long i, int E,
                                       int& s, int& t) {
  if (i < E) { s = src[i]; t = dst[i]; }
  else       { s = t = (int)(i - E); }            // self-loop
}

// ---------------- GAT layer 1: x[N,7] @ W1^T -> hlin[N,256] -------------
__global__ void k_lin1(const float* __restrict__ x, const float* __restrict__ W1,
                       float* __restrict__ hlin, int N) {
  long idx = (long)blockIdx.x * blockDim.x + threadIdx.x;
  if (idx >= (long)N * 256) return;
  int n = (int)(idx >> 8), oc = (int)(idx & 255);
  const float* xr = x + (size_t)n * 7;
  const float* wr = W1 + (size_t)oc * 7;
  float s = 0.f;
#pragma unroll
  for (int i = 0; i < 7; ++i) s += xr[i] * wr[i];
  hlin[idx] = s;
}

// a_src[n,h] = <hlin[n,h,:], att_src[h,:]> (D=64)
__global__ void k_att(const float* __restrict__ hlin, const float* __restrict__ att_s,
                      const float* __restrict__ att_d, float* __restrict__ aS,
                      float* __restrict__ aD, int N, int H) {
  int idx = blockIdx.x * blockDim.x + threadIdx.x;
  if (idx >= N * H) return;
  int n = idx / H, h = idx % H;
  const float* hr = hlin + ((size_t)n * H + h) * 64;
  const float* as = att_s + h * 64;
  const float* ad = att_d + h * 64;
  float ss = 0.f, sd = 0.f;
#pragma unroll 8
  for (int d = 0; d < 64; ++d) { float v = hr[d]; ss += v * as[d]; sd += v * ad[d]; }
  aS[idx] = ss; aD[idx] = sd;
}

__global__ void k_init(float* __restrict__ acc, long acc_n,
                       float* __restrict__ m, float* __restrict__ z, long mz_n) {
  long idx = (long)blockIdx.x * blockDim.x + threadIdx.x;
  if (idx < acc_n) acc[idx] = 0.f;
  if (idx < mz_n) { m[idx] = -__builtin_inff(); z[idx] = 0.f; }
}

// pass A: segment max of leaky(a_src[s]+a_dst[t]) per (dst, head)
__global__ void k_emax(const int* __restrict__ src, const int* __restrict__ dst,
                       const float* __restrict__ aS, const float* __restrict__ aD,
                       float* __restrict__ m, int E, int N, int H) {
  long idx = (long)blockIdx.x * blockDim.x + threadIdx.x;
  long tot = (long)(E + N) * H;
  if (idx >= tot) return;
  long i = idx / H; int h = (int)(idx % H);
  int s, t; get_st(src, dst, i, E, s, t);
  float e = lrelu_f(aS[(size_t)s * H + h] + aD[(size_t)t * H + h]);
  atomic_max_f(&m[(size_t)t * H + h], e);
}

// pass B: ex = exp(e - m[t]); z[t] += ex
__global__ void k_eexp(const int* __restrict__ src, const int* __restrict__ dst,
                       const float* __restrict__ aS, const float* __restrict__ aD,
                       const float* __restrict__ m, float* __restrict__ ex,
                       float* __restrict__ z, int E, int N, int H) {
  long idx = (long)blockIdx.x * blockDim.x + threadIdx.x;
  long tot = (long)(E + N) * H;
  if (idx >= tot) return;
  long i = idx / H; int h = (int)(idx % H);
  int s, t; get_st(src, dst, i, E, s, t);
  float e = lrelu_f(aS[(size_t)s * H + h] + aD[(size_t)t * H + h]);
  float v = __expf(e - m[(size_t)t * H + h]);
  ex[idx] = v;
  atomicAdd(&z[(size_t)t * H + h], v);
}

// pass C (float4): acc[t, d..d+3] += (ex/z[t]) * hlin[s, d..d+3]
__global__ void k_scatter4(const int* __restrict__ src, const int* __restrict__ dst,
                           const float* __restrict__ hlin, const float* __restrict__ ex,
                           const float* __restrict__ z, float* __restrict__ acc,
                           int E, int N, int H, int F) {
  int q = F >> 2;
  long idx = (long)blockIdx.x * blockDim.x + threadIdx.x;
  long tot = (long)(E + N) * q;
  if (idx >= tot) return;
  long i = idx / q; int d = (int)(idx % q) * 4; int h = d >> 6;
  int s, t; get_st(src, dst, i, E, s, t);
  float alpha = ex[i * H + h] / (z[(size_t)t * H + h] + 1e-16f);
  float4 hv = *(const float4*)(hlin + (size_t)s * F + d);
  float* ap = acc + (size_t)t * F + d;
  atomicAdd(ap + 0, alpha * hv.x);
  atomicAdd(ap + 1, alpha * hv.y);
  atomicAdd(ap + 2, alpha * hv.z);
  atomicAdd(ap + 3, alpha * hv.w);
}

// finalize: elu(bn(acc + bias)) -> f16
__global__ void k_fin(const float* __restrict__ acc, const float* __restrict__ bias,
                      const float* __restrict__ g, const float* __restrict__ b,
                      _Float16* __restrict__ o16, int N, int F) {
  long idx = (long)blockIdx.x * blockDim.x + threadIdx.x;
  if (idx >= (long)N * F) return;
  int c = (int)(idx % F);
  float v = acc[idx] + bias[c];
  v = v * (g[c] * rsqrtf(1.f + BN_EPS)) + b[c];
  o16[idx] = (_Float16)elu_f(v);
}

__global__ void k_cvt(const float* __restrict__ in, _Float16* __restrict__ o, long n) {
  long idx = (long)blockIdx.x * blockDim.x + threadIdx.x;
  if (idx < n) o[idx] = (_Float16)in[idx];
}

// -------- WMMA GEMM (layer 2): C[M,64] = A[M,256] x W[64,256]^T, f32 out --------
// One wave per M-tile; A fragment loaded once per K-step and reused across the
// 4 N-tiles (4 accumulators). Weights staged in LDS (ds_load fragments).
__global__ void k_gemm(const _Float16* __restrict__ A, const _Float16* __restrict__ Wg,
                       float* __restrict__ C, int M, int Ncol, int K) {
  __shared__ __attribute__((aligned(16))) _Float16 sW[64 * 256];
  stage_w(Wg, sW, Ncol * K);

  int wid  = (int)(((long)blockIdx.x * blockDim.x + threadIdx.x) >> 5);
  int lane = threadIdx.x & 31;
  int Mt = (M + 15) >> 4;
  if (wid >= Mt) return;
  int hi = (lane >> 4) & 1;
  int m  = min(wid * 16 + (lane & 15), M - 1);
  const _Float16* ar = A + (size_t)m * K + hi * 8;

  v8f c[4] = {};
  for (int kb = 0; kb < K; kb += 32) {
    v16h a = ld_frag(ar + kb, 16);
#pragma unroll
    for (int nt = 0; nt < 4; ++nt) {
      const _Float16* wr = sW + (size_t)(nt * 16 + (lane & 15)) * K + hi * 8;
      v16h bb = ld_frag(wr + kb, 16);
      c[nt] = WMMA_F16(a, bb, c[nt]);
    }
  }
  int row0 = wid * 16 + hi * 8;
#pragma unroll
  for (int nt = 0; nt < 4; ++nt) {
    int col = nt * 16 + (lane & 15);
#pragma unroll
    for (int r = 0; r < 8; ++r) {
      int row = row0 + r;
      if (row < M) C[(size_t)row * Ncol + col] = c[nt][r];
    }
  }
}

// -------- edge feature MLP: ef = elu(edge_attr @ et_W^T + et_b) -> f16 [E,32] ----
__global__ void k_ef(const float* __restrict__ ea, const float* __restrict__ etW,
                     const float* __restrict__ etb, _Float16* __restrict__ ef, int E) {
  long idx = (long)blockIdx.x * blockDim.x + threadIdx.x;
  if (idx >= (long)E * 32) return;
  int e = (int)(idx >> 5), j = (int)(idx & 31);
  const float* er = ea + (size_t)e * 8;
  const float* wr = etW + (size_t)j * 8;
  float s = etb[j];
#pragma unroll
  for (int i = 0; i < 8; ++i) s += er[i] * wr[i];
  ef[idx] = (_Float16)elu_f(s);
}

// -------- c1: z1 = bn(elu([h[src],h[dst],ef] @ c1_W^T + c1_b)) -> f16 [E,64] -----
// One wave per M-tile: the 5 gathered A fragments (h2[src], h2[dst], ef) are
// loaded once and reused across all 4 N-tiles. Weights (64x160) staged in LDS.
__global__ void k_c1(const int* __restrict__ src, const int* __restrict__ dst,
                     const _Float16* __restrict__ h2, const _Float16* __restrict__ ef,
                     const _Float16* __restrict__ Wg, const float* __restrict__ bias,
                     const float* __restrict__ g, const float* __restrict__ b,
                     _Float16* __restrict__ o, int E) {
  __shared__ __attribute__((aligned(16))) _Float16 sW[64 * 160];
  stage_w(Wg, sW, 64 * 160);

  int wid  = (int)(((long)blockIdx.x * blockDim.x + threadIdx.x) >> 5);
  int lane = threadIdx.x & 31;
  int Mt = (E + 15) >> 4;
  if (wid >= Mt) return;
  int hi = (lane >> 4) & 1;
  int e  = min(wid * 16 + (lane & 15), E - 1);
  int s = src[e], t = dst[e];
  const _Float16* p0 = h2 + (size_t)s * 64 + hi * 8;
  const _Float16* p1 = h2 + (size_t)t * 64 + hi * 8;
  const _Float16* p2 = ef + (size_t)e * 32 + hi * 8;

  // gather A fragments once (K = 160 in five 32-chunks)
  v16h a0 = ld_frag(p0,      16);
  v16h a1 = ld_frag(p0 + 32, 16);
  v16h a2 = ld_frag(p1,      16);
  v16h a3 = ld_frag(p1 + 32, 16);
  v16h a4 = ld_frag(p2,      16);

  int row0 = wid * 16 + hi * 8;
#pragma unroll
  for (int nt = 0; nt < 4; ++nt) {
    const _Float16* wr = sW + (size_t)(nt * 16 + (lane & 15)) * 160 + hi * 8;
    v8f c = {};
    c = WMMA_F16(a0, ld_frag(wr,       16), c);
    c = WMMA_F16(a1, ld_frag(wr + 32,  16), c);
    c = WMMA_F16(a2, ld_frag(wr + 64,  16), c);
    c = WMMA_F16(a3, ld_frag(wr + 96,  16), c);
    c = WMMA_F16(a4, ld_frag(wr + 128, 16), c);
    int col = nt * 16 + (lane & 15);
    float sc = g[col] * rsqrtf(1.f + BN_EPS);
    float bc = b[col], bi = bias[col];
#pragma unroll
    for (int r = 0; r < 8; ++r) {
      int row = row0 + r;
      if (row < E) {
        float v = elu_f(c[r] + bi);        // elu BEFORE bn (reference order)
        v = v * sc + bc;
        o[(size_t)row * 64 + col] = (_Float16)v;
      }
    }
  }
}

// -------- c2: z2 = elu(z1 @ c2_W^T + c2_b) -> f16 [E,32] ------------------------
// One wave per M-tile, 2 N-tiles reusing the two A fragments. Weights in LDS.
__global__ void k_c2(const _Float16* __restrict__ A, const _Float16* __restrict__ Wg,
                     const float* __restrict__ bias, _Float16* __restrict__ o, int M) {
  __shared__ __attribute__((aligned(16))) _Float16 sW[32 * 64];
  stage_w(Wg, sW, 32 * 64);

  int wid  = (int)(((long)blockIdx.x * blockDim.x + threadIdx.x) >> 5);
  int lane = threadIdx.x & 31;
  int Mt = (M + 15) >> 4;
  if (wid >= Mt) return;
  int hi = (lane >> 4) & 1;
  int m  = min(wid * 16 + (lane & 15), M - 1);
  const _Float16* ar = A + (size_t)m * 64 + hi * 8;
  v16h a0 = ld_frag(ar,      16);
  v16h a1 = ld_frag(ar + 32, 16);

  int row0 = wid * 16 + hi * 8;
#pragma unroll
  for (int nt = 0; nt < 2; ++nt) {
    const _Float16* wr = sW + (size_t)(nt * 16 + (lane & 15)) * 64 + hi * 8;
    v8f c = {};
    c = WMMA_F16(a0, ld_frag(wr,      16), c);
    c = WMMA_F16(a1, ld_frag(wr + 32, 16), c);
    int col = nt * 16 + (lane & 15);
    float bi = bias[col];
#pragma unroll
    for (int r = 0; r < 8; ++r) {
      int row = row0 + r;
      if (row < M) o[(size_t)row * 32 + col] = (_Float16)elu_f(c[r] + bi);
    }
  }
}

// -------- final: sigmoid(z2 @ c3_W^T + c3_b) ------------------------------------
__global__ void k_out(const _Float16* __restrict__ z2, const float* __restrict__ w,
                      const float* __restrict__ b, float* __restrict__ out, int E) {
  long idx = (long)blockIdx.x * blockDim.x + threadIdx.x;
  if (idx >= E) return;
  const _Float16* zr = z2 + idx * 32;
  float s = b[0];
#pragma unroll
  for (int j = 0; j < 32; ++j) s += (float)zr[j] * w[j];
  out[idx] = 1.f / (1.f + __expf(-s));
}

extern "C" void kernel_launch(void* const* d_in, const int* in_sizes, int n_in,
                              void* d_out, int out_size, void* d_ws, size_t ws_size,
                              hipStream_t stream) {
  const float* x    = (const float*)d_in[0];
  const int*   ei   = (const int*)d_in[1];
  const float* ea   = (const float*)d_in[2];
  const float* W1   = (const float*)d_in[3];
  const float* as1  = (const float*)d_in[4];
  const float* ad1  = (const float*)d_in[5];
  const float* b1   = (const float*)d_in[6];
  const float* W2   = (const float*)d_in[7];
  const float* as2  = (const float*)d_in[8];
  const float* ad2  = (const float*)d_in[9];
  const float* b2   = (const float*)d_in[10];
  const float* bn1g = (const float*)d_in[11];
  const float* bn1b = (const float*)d_in[12];
  const float* bn2g = (const float*)d_in[13];
  const float* bn2b = (const float*)d_in[14];
  const float* etW  = (const float*)d_in[15];
  const float* etb  = (const float*)d_in[16];
  const float* c1W  = (const float*)d_in[17];
  const float* c1b  = (const float*)d_in[18];
  const float* bncg = (const float*)d_in[19];
  const float* bncb = (const float*)d_in[20];
  const float* c2W  = (const float*)d_in[21];
  const float* c2b  = (const float*)d_in[22];
  const float* c3W  = (const float*)d_in[23];
  const float* c3b  = (const float*)d_in[24];
  float* out = (float*)d_out;

  int N = in_sizes[0] / 7;
  int E = in_sizes[1] / 2;
  const int* src = ei;
  const int* dst = ei + E;
  long ET = (long)E + N;

  // --- carve workspace (256B aligned) ---
  char* p = (char*)d_ws;
  auto alloc = [&](size_t bytes) -> void* {
    void* r = (void*)p;
    p += (bytes + 255) & ~(size_t)255;
    return r;
  };
  float*     hlin1 = (float*)alloc((size_t)N * 256 * 4);
  float*     aS1   = (float*)alloc((size_t)N * 4 * 4);
  float*     aD1   = (float*)alloc((size_t)N * 4 * 4);
  float*     m1    = (float*)alloc((size_t)N * 4 * 4);
  float*     zs1   = (float*)alloc((size_t)N * 4 * 4);
  float*     ex1   = (float*)alloc((size_t)ET * 4 * 4);
  float*     acc1  = (float*)alloc((size_t)N * 256 * 4);
  _Float16*  h1h   = (_Float16*)alloc((size_t)N * 256 * 2);
  _Float16*  W2h   = (_Float16*)alloc((size_t)64 * 256 * 2);
  float*     h2lin = (float*)alloc((size_t)N * 64 * 4);
  float*     aS2   = (float*)alloc((size_t)N * 4);
  float*     aD2   = (float*)alloc((size_t)N * 4);
  float*     m2    = (float*)alloc((size_t)N * 4);
  float*     zs2   = (float*)alloc((size_t)N * 4);
  float*     ex2   = (float*)alloc((size_t)ET * 4);
  float*     acc2  = (float*)alloc((size_t)N * 64 * 4);
  _Float16*  h2h   = (_Float16*)alloc((size_t)N * 64 * 2);
  _Float16*  efh   = (_Float16*)alloc((size_t)E * 32 * 2);
  _Float16*  c1Wh  = (_Float16*)alloc((size_t)64 * 160 * 2);
  _Float16*  c2Wh  = (_Float16*)alloc((size_t)32 * 64 * 2);
  _Float16*  z1h   = (_Float16*)alloc((size_t)E * 64 * 2);
  _Float16*  z2h   = (_Float16*)alloc((size_t)E * 32 * 2);

  const int B = 256;
  auto nb = [](long n, int b) { return (int)((n + b - 1) / b); };

  // ---- GAT layer 1 ----
  k_lin1<<<nb((long)N * 256, B), B, 0, stream>>>(x, W1, hlin1, N);
  k_att<<<nb((long)N * 4, B), B, 0, stream>>>(hlin1, as1, ad1, aS1, aD1, N, 4);
  k_init<<<nb((long)N * 256, B), B, 0, stream>>>(acc1, (long)N * 256, m1, zs1, (long)N * 4);
  k_emax<<<nb(ET * 4, B), B, 0, stream>>>(src, dst, aS1, aD1, m1, E, N, 4);
  k_eexp<<<nb(ET * 4, B), B, 0, stream>>>(src, dst, aS1, aD1, m1, ex1, zs1, E, N, 4);
  k_scatter4<<<nb(ET * 64, B), B, 0, stream>>>(src, dst, hlin1, ex1, zs1, acc1, E, N, 4, 256);
  k_fin<<<nb((long)N * 256, B), B, 0, stream>>>(acc1, b1, bn1g, bn1b, h1h, N, 256);

  // ---- GAT layer 2 (linear via WMMA, then attention) ----
  k_cvt<<<nb(64 * 256, B), B, 0, stream>>>(W2, W2h, 64 * 256);
  long t2 = (long)((N + 15) / 16);
  k_gemm<<<nb(t2 * 32, B), B, 0, stream>>>(h1h, W2h, h2lin, N, 64, 256);
  k_att<<<nb((long)N, B), B, 0, stream>>>(h2lin, as2, ad2, aS2, aD2, N, 1);
  k_init<<<nb((long)N * 64, B), B, 0, stream>>>(acc2, (long)N * 64, m2, zs2, (long)N);
  k_emax<<<nb(ET, B), B, 0, stream>>>(src, dst, aS2, aD2, m2, E, N, 1);
  k_eexp<<<nb(ET, B), B, 0, stream>>>(src, dst, aS2, aD2, m2, ex2, zs2, E, N, 1);
  k_scatter4<<<nb(ET * 16, B), B, 0, stream>>>(src, dst, h2lin, ex2, zs2, acc2, E, N, 1, 64);
  k_fin<<<nb((long)N * 64, B), B, 0, stream>>>(acc2, b2, bn2g, bn2b, h2h, N, 64);

  // ---- edge classifier ----
  k_ef<<<nb((long)E * 32, B), B, 0, stream>>>(ea, etW, etb, efh, E);
  k_cvt<<<nb(64 * 160, B), B, 0, stream>>>(c1W, c1Wh, 64 * 160);
  k_cvt<<<nb(32 * 64, B), B, 0, stream>>>(c2W, c2Wh, 32 * 64);
  long tc = (long)((E + 15) / 16);
  k_c1<<<nb(tc * 32, B), B, 0, stream>>>(src, dst, h2h, efh, c1Wh, c1b, bncg, bncb, z1h, E);
  k_c2<<<nb(tc * 32, B), B, 0, stream>>>(z1h, c2Wh, c2b, z2h, E);
  k_out<<<nb((long)E, B), B, 0, stream>>>(z2h, c3W, c3b, out, E);
}
